// BitLinear_48060684042971
// MI455X (gfx1250) — 8-bit matrix operand order repaired and checked by reading, compiled, not run on hardware
//
#include <hip/hip_runtime.h>
#include <hip/hip_bf16.h>
#include <stdint.h>


typedef int v8i __attribute__((ext_vector_type(8)));

constexpr int M_TOT = 4 * 4096;
constexpr int N_TOT = 2048;
constexpr int K_TOT = 2048;
constexpr int KC    = K_TOT / 64;
constexpr float Q_MAX = 127.0f;

constexpr size_t WS_PARTIAL = 0;
constexpr size_t WS_SCAL    = 4096;
constexpr size_t WS_ASCALE  = 8192;
constexpr size_t WS_XQ      = 131072;
constexpr size_t WS_WQ      = WS_XQ + (size_t)M_TOT * K_TOT;

__global__ void k_wabs_partial(const float* __restrict__ w, float* __restrict__ partial, int n) {
    __shared__ float sm[256];
    float s = 0.0f;
    for (int i = blockIdx.x * blockDim.x + threadIdx.x; i < n; i += gridDim.x * blockDim.x)
        s += fabsf(w[i]);
    sm[threadIdx.x] = s;
    __syncthreads();
    for (int off = 128; off > 0; off >>= 1) {
        if ((int)threadIdx.x < off) sm[threadIdx.x] += sm[threadIdx.x + off];
        __syncthreads();
    }
    if (threadIdx.x == 0) partial[blockIdx.x] = sm[0];
}

__global__ void k_wscale(const float* __restrict__ partial, float* __restrict__ scal, float inv_count) {
    __shared__ float sm[256];
    sm[threadIdx.x] = partial[threadIdx.x];
    __syncthreads();
    for (int off = 128; off > 0; off >>= 1) {
        if ((int)threadIdx.x < off) sm[threadIdx.x] += sm[threadIdx.x + off];
        __syncthreads();
    }
    if (threadIdx.x == 0) {
        float ws = fmaxf(sm[0] * inv_count, 1e-6f);
        scal[0] = ws;
        scal[1] = 1.0f / ws;
    }
}

__device__ __forceinline__ int quant_clip(float v, float q, float lim) {
    float t = rintf(v * q);
    t = fminf(fmaxf(t, -lim), lim);
    return (int)t;
}
__device__ __forceinline__ int pack4(float4 f, float q, float lim) {
    int a = quant_clip(f.x, q, lim) & 0xFF;
    int b = quant_clip(f.y, q, lim) & 0xFF;
    int c = quant_clip(f.z, q, lim) & 0xFF;
    int d = quant_clip(f.w, q, lim) & 0xFF;
    return a | (b << 8) | (c << 16) | (d << 24);
}

__global__ __launch_bounds__(256)
void k_quant_x(const float* __restrict__ x, float* __restrict__ a_scale,
               v8i* __restrict__ xqf) {
    __shared__ float sm[256];
    __shared__ float sq[16];

    const int tid  = threadIdx.x;
    const int lane = tid & 31;
    const int wave = tid >> 5;
    const int mp   = blockIdx.x;
    const int half = lane >> 4;
    const int row  = mp * 16 + (lane & 15);
    const float4* xr = (const float4*)(x + (size_t)row * K_TOT);

    float4 vals[4][8];
    float m = 0.0f;
#pragma unroll
    for (int t = 0; t < 4; ++t) {
        const int kc = wave * 4 + t;
#pragma unroll
        for (int v = 0; v < 8; ++v) {
            const int c = ((v >> 1) << 2) + (half << 1) + (v & 1);
            float4 f = xr[kc * 16 + c];
            vals[t][v] = f;
            m = fmaxf(m, fmaxf(fmaxf(fabsf(f.x), fabsf(f.y)),
                               fmaxf(fabsf(f.z), fabsf(f.w))));
        }
    }

    sm[tid] = m;
    __syncthreads();
    if (tid < 16) {
        float rm = 0.0f;
#pragma unroll
        for (int i = 0; i < 16; ++i) rm = fmaxf(rm, sm[tid + 16 * i]);
        rm = fmaxf(rm, 1e-8f);
        a_scale[mp * 16 + tid] = rm;
        sq[tid] = Q_MAX / rm;
    }
    __syncthreads();

    const float q = sq[lane & 15];
#pragma unroll
    for (int t = 0; t < 4; ++t) {
        const int kc = wave * 4 + t;
        v8i out;
#pragma unroll
        for (int v = 0; v < 8; ++v) out[v] = pack4(vals[t][v], q, 127.0f);
        xqf[(size_t)(mp * KC + kc) * 32 + lane] = out;
    }
}

__global__ void k_quant_w(const float* __restrict__ w, const float* __restrict__ scal,
                          v8i* __restrict__ wqf) {
    const int lane = threadIdx.x & 31;
    const int wave = threadIdx.x >> 5;
    const int kc   = blockIdx.x * 8 + wave;
    const int np   = blockIdx.y;
    const int half = lane >> 4;
    const int row  = np * 16 + (lane & 15);
    const float inv_ws = scal[1];
    const float* wr = w + (size_t)row * K_TOT + kc * 64;
    v8i out;
#pragma unroll
    for (int v = 0; v < 8; ++v) {
        const int kstart = ((v >> 1) << 4) + (half << 3) + ((v & 1) << 2);
        float4 f = *(const float4*)(wr + kstart);
        out[v] = pack4(f, inv_ws, 1.0f);
    }
    wqf[(size_t)(np * KC + kc) * 32 + lane] = out;
}

__global__ __launch_bounds__(256)
void k_gemm(const v8i* __restrict__ xqf, const v8i* __restrict__ wqf,
            const float* __restrict__ a_scale, const float* __restrict__ scal,
            float* __restrict__ y) {
    const int lane = threadIdx.x & 31;
    const int wave = threadIdx.x >> 5;
    const int wm   = wave >> 1;
    const int wn   = wave & 1;
    const int m_base = blockIdx.y * 256 + wm * 64;
    const int n_base = blockIdx.x * 256 + wn * 128;

    const v8i* abase = xqf + (size_t)(m_base >> 4) * KC * 32 + lane;
    const v8i* bbase = wqf + (size_t)(n_base >> 4) * KC * 32 + lane;

    v8i acc[4][8] = {};

#pragma unroll 2
    for (int kc = 0; kc < KC; ++kc) {
        v8i A[4], B[8];
#pragma unroll
        for (int i = 0; i < 4; ++i) A[i] = abase[(size_t)i * (KC * 32) + kc * 32];
#pragma unroll
        for (int j = 0; j < 8; ++j) B[j] = bbase[(size_t)j * (KC * 32) + kc * 32];

        if (kc + 1 < KC) {
            __builtin_prefetch((const void*)(abase + (kc + 1) * 32), 0, 3);
            __builtin_prefetch((const void*)(abase + 2 * (KC * 32) + (kc + 1) * 32), 0, 3);
            __builtin_prefetch((const void*)(bbase + (kc + 1) * 32), 0, 3);
            __builtin_prefetch((const void*)(bbase + 4 * (KC * 32) + (kc + 1) * 32), 0, 3);
        }

#pragma unroll
        for (int i = 0; i < 4; ++i)
#pragma unroll
            for (int j = 0; j < 8; ++j)
                acc[i][j] = __builtin_amdgcn_wmma_i32_16x16x64_iu8(
                     true, A[i],  true, B[j], acc[i][j],
                     false,  false);
    }

    const float s127 = scal[0] * (1.0f / 127.0f);
    const int half  = lane >> 4;
    const int nlane = lane & 15;
#pragma unroll
    for (int i = 0; i < 4; ++i) {
        const int rbase = m_base + i * 16 + half * 8;
        float as[8];
#pragma unroll
        for (int v = 0; v < 8; ++v) as[v] = a_scale[rbase + v] * s127;
#pragma unroll
        for (int j = 0; j < 8; ++j) {
            const int n = n_base + j * 16 + nlane;
#pragma unroll
            for (int v = 0; v < 8; ++v)
                y[(size_t)(rbase + v) * N_TOT + n] = (float)acc[i][j][v] * as[v];
        }
    }
}

extern "C" void kernel_launch(void* const* d_in, const int* in_sizes, int n_in,
                              void* d_out, int out_size, void* d_ws, size_t ws_size,
                              hipStream_t stream) {
    (void)in_sizes; (void)n_in; (void)out_size; (void)ws_size;
    const float* x = (const float*)d_in[0];
    const float* w = (const float*)d_in[1];
    float* y = (float*)d_out;

    char* ws = (char*)d_ws;
    float* partial = (float*)(ws + WS_PARTIAL);
    float* scal    = (float*)(ws + WS_SCAL);
    float* ascale  = (float*)(ws + WS_ASCALE);
    v8i*   xqf     = (v8i*)(ws + WS_XQ);
    v8i*   wqf     = (v8i*)(ws + WS_WQ);

    k_wabs_partial<<<256, 256, 0, stream>>>(w, partial, N_TOT * K_TOT);
    k_wscale<<<1, 256, 0, stream>>>(partial, scal, 1.0f / (float)(N_TOT * K_TOT));
    k_quant_w<<<dim3(4, N_TOT / 16), 256, 0, stream>>>(w, scal, wqf);
    k_quant_x<<<M_TOT / 16, 256, 0, stream>>>(x, ascale, xqf);
    k_gemm<<<dim3(N_TOT / 256, M_TOT / 256), 256, 0, stream>>>(xqf, wqf, ascale, scal, y);
}
